// CharAutoencoder_13365938225894
// MI455X (gfx1250) — matrix operand order repaired and checked by reading, not compiled, not run on hardware
//
#include <hip/hip_runtime.h>
#include <hip/hip_bf16.h>

typedef __bf16 bf16_t;
typedef bf16_t v16bf __attribute__((ext_vector_type(16)));
typedef bf16_t v8bf  __attribute__((ext_vector_type(8)));
typedef float  v8f   __attribute__((ext_vector_type(8)));
typedef int    v4i   __attribute__((ext_vector_type(4)));

static constexpr int Bn = 256;   // batch
static constexpr int Sn = 128;   // seq len
static constexpr int En = 256;   // embed
static constexpr int Hn = 1024;  // hidden
static constexpr int Ln = 512;   // latent
static constexpr int Vn = 30;    // vocab+1
static constexpr int G4 = 4 * Hn;   // 4096 gates
static constexpr int KE = En + Hn;  // 1280 fused K for encoder

// ---- gfx1250 async global->LDS path (probe-guarded; fallback keeps compile green) ----
#define USE_ASYNC 0
#if defined(__has_builtin)
#if __has_builtin(__builtin_amdgcn_global_load_async_to_lds_b128) && \
    __has_builtin(__builtin_amdgcn_s_wait_asynccnt)
#undef USE_ASYNC
#define USE_ASYNC 1
#endif
#endif

#define AS1 __attribute__((address_space(1)))
#define AS3 __attribute__((address_space(3)))

// ---------------- device helpers ----------------

__device__ __forceinline__ float sigf(float x) { return 1.0f / (1.0f + __expf(-x)); }

// 16-bit A fragment 16x32 (MxK): lane L row M=L&15 holds K={klo..klo+15}, klo=(L>=16)*16
__device__ __forceinline__ v16bf load_afrag(const bf16_t* __restrict__ row, int kc, int lane) {
    int klo = (lane >> 4) * 16;
    v8bf a0 = *(const v8bf*)(row + kc + klo);
    v8bf a1 = *(const v8bf*)(row + kc + 8 + klo);
    return __builtin_shufflevector(a0, a1, 0,1,2,3,4,5,6,7,8,9,10,11,12,13,14,15);
}

// Stage one 128-col x 32-k B tile (8KB) into LDS; 256 threads, 32B each.
__device__ __forceinline__ void stage_b(const bf16_t* __restrict__ Bw, int ldb, int n0, int kc,
                                        bf16_t* __restrict__ dst, int tid) {
    int r  = tid >> 1;          // 0..127 rows (output columns)
    int co = (tid & 1) * 16;    // element offset within 32-k row
    const bf16_t* src = Bw + (size_t)(n0 + r) * ldb + kc + co;
    bf16_t* d = dst + r * 32 + co;
#if USE_ASYNC
    __builtin_amdgcn_global_load_async_to_lds_b128(
        (v4i*)(void*)src, (AS3 v4i*)(AS3 void*)d, 0, 0);
    __builtin_amdgcn_global_load_async_to_lds_b128(
        (v4i*)(void*)(src + 8), (AS3 v4i*)(AS3 void*)(d + 8), 0, 0);
#else
    *(v8bf*)d = *(const v8bf*)src;
    *(v8bf*)(d + 8) = *(const v8bf*)(src + 8);
#endif
}

__device__ __forceinline__ void stage_wait() {
#if USE_ASYNC
    __builtin_amdgcn_s_wait_asynccnt(0);
#endif
    __syncthreads();
}

// B fragment 32x16 (KxN) from LDS tile: lane holds column N=lane&15 (+16j), 16 contiguous K at (lane>>4)*16
__device__ __forceinline__ v16bf lds_bfrag(const bf16_t* __restrict__ sb, int j, int lane) {
    return *(const v16bf*)(sb + ((size_t)(j * 16 + (lane & 15)) * 32) + ((lane >> 4) * 16));
}

// ---------------- prep kernels ----------------

__global__ void pack_concat(const float* __restrict__ wih, const float* __restrict__ whh,
                            bf16_t* __restrict__ dst, int N, int K1, int K2) {
    size_t g = (size_t)blockIdx.x * blockDim.x + threadIdx.x;
    int K = K1 + K2;
    if (g >= (size_t)N * K) return;
    int n = (int)(g / K), k = (int)(g % K);
    float v = (k < K1) ? wih[(size_t)n * K1 + k] : whh[(size_t)n * K2 + (k - K1)];
    dst[g] = (bf16_t)v;
}

__global__ void pack_strided(const float* __restrict__ src, bf16_t* __restrict__ dst,
                             int N, int K, int srcld, int srcoff, int Nvalid) {
    size_t g = (size_t)blockIdx.x * blockDim.x + threadIdx.x;
    if (g >= (size_t)N * K) return;
    int n = (int)(g / K), k = (int)(g % K);
    float v = (n < Nvalid) ? src[(size_t)n * srcld + srcoff + k] : 0.0f;
    dst[g] = (bf16_t)v;
}

__global__ void add2(const float* __restrict__ a, const float* __restrict__ b,
                     float* __restrict__ dst, int n) {
    int i = blockIdx.x * blockDim.x + threadIdx.x;
    if (i < n) dst[i] = a[i] + b[i];
}

__global__ void zero_f32(float* __restrict__ p, size_t n) {
    size_t i = (size_t)blockIdx.x * blockDim.x + threadIdx.x;
    if (i < n) p[i] = 0.0f;
}

__global__ void zero_bf16(bf16_t* __restrict__ p, size_t n) {
    size_t i = (size_t)blockIdx.x * blockDim.x + threadIdx.x;
    if (i < n) p[i] = (bf16_t)0.0f;
}

// embedded[s][b][k] = emb[x[b][s]][k] as bf16
__global__ void embed_gather(const int* __restrict__ x, const float* __restrict__ emb,
                             bf16_t* __restrict__ Xb) {
    size_t g = (size_t)blockIdx.x * blockDim.x + threadIdx.x;
    if (g >= (size_t)Sn * Bn * En) return;
    int k = (int)(g % En);
    size_t t = g / En;
    int b = (int)(t % Bn);
    int s = (int)(t / Bn);
    int tok = x[(size_t)b * Sn + s];
    Xb[g] = (bf16_t)emb[(size_t)tok * En + k];
}

// ---------------- encoder step GEMM: G[dir] = [x_t, h_prev] @ [Wih;Whh]^T + bias ----------------
// grid (G4/128, Bn/128, 2 dirs), 256 threads = 8 waves, wave tile 16x128 (8 accumulators)

__global__ __launch_bounds__(256)
void enc_step_gemm(const bf16_t* __restrict__ Xb,     // [S][B][E]
                   const bf16_t* __restrict__ Wp,     // [2][4096][1280]
                   const bf16_t* __restrict__ Hprev,  // [2][B][H]
                   const float*  __restrict__ biasE,  // [2][4096]
                   float* __restrict__ G,             // [2][B][4096]
                   int t) {
    __shared__ bf16_t sB[2][128 * 32];
    const int dir  = blockIdx.z;
    const int n0   = blockIdx.x * 128;
    const int m0   = blockIdx.y * 128;
    const int tid  = threadIdx.x;
    const int wave = tid >> 5;
    const int lane = tid & 31;
    const int trow = dir ? (Sn - 1 - t) : t;
    const int mrow = m0 + wave * 16 + (lane & 15);

    const bf16_t* Ax = Xb + ((size_t)trow * Bn + mrow) * En;
    const bf16_t* Ah = Hprev + ((size_t)dir * Bn + mrow) * Hn;
    const bf16_t* W  = Wp + (size_t)dir * G4 * KE;

    v8f acc[8] = { {}, {}, {}, {}, {}, {}, {}, {} };
    const int nc = KE / 32;
    stage_b(W, KE, n0, 0, sB[0], tid);
    stage_wait();
    for (int c = 0; c < nc; ++c) {
        int kc = c * 32;
        if (c + 1 < nc) stage_b(W, KE, n0, kc + 32, sB[(c + 1) & 1], tid);
        v16bf a = (kc < En) ? load_afrag(Ax, kc, lane) : load_afrag(Ah, kc - En, lane);
        const bf16_t* sb = sB[c & 1];
        #pragma unroll
        for (int j = 0; j < 8; ++j) {
            v16bf b = lds_bfrag(sb, j, lane);
            acc[j] = __builtin_amdgcn_wmma_f32_16x16x32_bf16(
                false, a, false, b, (short)0, acc[j], false, false);
        }
        stage_wait();
    }
    float* Gd = G + (size_t)dir * Bn * G4;
    const float* bs = biasE + (size_t)dir * G4;
    #pragma unroll
    for (int j = 0; j < 8; ++j) {
        int n = n0 + j * 16 + (lane & 15);
        float bv = bs[n];
        #pragma unroll
        for (int i = 0; i < 8; ++i) {
            int m = m0 + wave * 16 + ((lane >> 4) << 3) + i;
            Gd[(size_t)m * G4 + n] = acc[j][i] + bv;
        }
    }
}

// ---------------- generic WMMA GEMM: C = A @ B^T (+addend) (+bias), optional bf16 mirror ----------------
// A: [M][lda] bf16, B: [N][ldb] bf16 row-major, grid (N/128, M/128)

__global__ __launch_bounds__(256)
void gemm_bf16_wmma(const bf16_t* __restrict__ A, int lda,
                    const bf16_t* __restrict__ Bw, int ldb,
                    float* __restrict__ Cout, int ldc,
                    const float* __restrict__ addend,     // nullable, stride ldc
                    const float* __restrict__ bias,       // nullable, [N]
                    bf16_t* __restrict__ mirror, int ldm, // nullable
                    int K, int Nvalid) {
    __shared__ bf16_t sB[2][128 * 32];
    const int n0   = blockIdx.x * 128;
    const int m0   = blockIdx.y * 128;
    const int tid  = threadIdx.x;
    const int wave = tid >> 5;
    const int lane = tid & 31;
    const int mrow = m0 + wave * 16 + (lane & 15);
    const bf16_t* Arow = A + (size_t)mrow * lda;

    v8f acc[8] = { {}, {}, {}, {}, {}, {}, {}, {} };
    const int nc = K / 32;
    stage_b(Bw, ldb, n0, 0, sB[0], tid);
    stage_wait();
    for (int c = 0; c < nc; ++c) {
        int kc = c * 32;
        if (c + 1 < nc) stage_b(Bw, ldb, n0, kc + 32, sB[(c + 1) & 1], tid);
        v16bf a = load_afrag(Arow, kc, lane);
        const bf16_t* sb = sB[c & 1];
        #pragma unroll
        for (int j = 0; j < 8; ++j) {
            v16bf b = lds_bfrag(sb, j, lane);
            acc[j] = __builtin_amdgcn_wmma_f32_16x16x32_bf16(
                false, a, false, b, (short)0, acc[j], false, false);
        }
        stage_wait();
    }
    #pragma unroll
    for (int j = 0; j < 8; ++j) {
        int n = n0 + j * 16 + (lane & 15);
        if (n >= Nvalid) continue;
        float bv = bias ? bias[n] : 0.0f;
        #pragma unroll
        for (int i = 0; i < 8; ++i) {
            int m = m0 + wave * 16 + ((lane >> 4) << 3) + i;
            size_t ci = (size_t)m * ldc + n;
            float v = acc[j][i] + bv;
            if (addend) v += addend[ci];
            Cout[ci] = v;
            if (mirror) mirror[(size_t)m * ldm + n] = (bf16_t)v;
        }
    }
}

// ---------------- LSTM pointwise: gate nonlinearities + state update ----------------
// grid (Bn*Hn/256, nz). Torch gate order i,f,g,o in 1024-wide chunks of G.

__global__ void lstm_pointwise(const float* __restrict__ G, size_t gz,
                               float* __restrict__ C, size_t cz,
                               bf16_t* __restrict__ Hout, size_t hz,
                               bf16_t* __restrict__ Hrec, size_t rz, int rld) {
    int z   = blockIdx.y;
    int idx = blockIdx.x * blockDim.x + threadIdx.x;  // b*Hn + hh
    int b   = idx >> 10;
    int hh  = idx & (Hn - 1);
    const float* g = G + z * gz + (size_t)b * G4;
    float i_ = g[hh], f_ = g[Hn + hh], gg = g[2 * Hn + hh], o_ = g[3 * Hn + hh];
    float c = C[z * cz + idx];
    c = sigf(f_) * c + sigf(i_) * tanhf(gg);
    float h = sigf(o_) * tanhf(c);
    C[z * cz + idx] = c;
    Hout[z * hz + idx] = (bf16_t)h;
    if (Hrec) Hrec[z * rz + (size_t)b * rld + hh] = (bf16_t)h;
}

// ---------------- host orchestration ----------------

extern "C" void kernel_launch(void* const* d_in, const int* in_sizes, int n_in,
                              void* d_out, int out_size, void* d_ws, size_t ws_size,
                              hipStream_t stream) {
    (void)in_sizes; (void)n_in; (void)out_size; (void)ws_size;

    const int*   x       = (const int*)  d_in[0];
    const float* emb     = (const float*)d_in[1];
    const float* e_wih_f = (const float*)d_in[2];
    const float* e_whh_f = (const float*)d_in[3];
    const float* e_bih_f = (const float*)d_in[4];
    const float* e_bhh_f = (const float*)d_in[5];
    const float* e_wih_b = (const float*)d_in[6];
    const float* e_whh_b = (const float*)d_in[7];
    const float* e_bih_b = (const float*)d_in[8];
    const float* e_bhh_b = (const float*)d_in[9];
    const float* enc_w   = (const float*)d_in[10];
    const float* enc_b   = (const float*)d_in[11];
    const float* dec_w   = (const float*)d_in[12];
    const float* dec_b   = (const float*)d_in[13];
    const float* d_wih   = (const float*)d_in[14];
    const float* d_whh   = (const float*)d_in[15];
    const float* d_bih   = (const float*)d_in[16];
    const float* d_bhh   = (const float*)d_in[17];
    const float* out_w   = (const float*)d_in[18];
    const float* out_b   = (const float*)d_in[19];
    float* outF = (float*)d_out;

    // ---- workspace bump allocator (~80 MB total) ----
    char* ws = (char*)d_ws;
    size_t off = 0;
    auto alloc = [&](size_t bytes) -> void* {
        void* p = ws + off;
        off = (off + bytes + 255) & ~(size_t)255;
        return p;
    };
    bf16_t* Xb    = (bf16_t*)alloc((size_t)Sn * Bn * En * 2);       // 16.8 MB
    bf16_t* Wp    = (bf16_t*)alloc((size_t)2 * G4 * KE * 2);        // 21.0 MB
    bf16_t* Wdhh  = (bf16_t*)alloc((size_t)G4 * Hn * 2);            //  8.4 MB
    bf16_t* Wdih  = (bf16_t*)alloc((size_t)G4 * Hn * 2);            //  8.4 MB
    bf16_t* Wenc  = (bf16_t*)alloc((size_t)Ln * 2 * Hn * 2);        //  2.1 MB
    bf16_t* Wdec  = (bf16_t*)alloc((size_t)Hn * Ln * 2);            //  1.0 MB
    bf16_t* Wout  = (bf16_t*)alloc((size_t)128 * Hn * 2);           //  0.26 MB (padded to BN)
    float*  biasE = (float*) alloc((size_t)2 * G4 * 4);
    float*  biasD = (float*) alloc((size_t)G4 * 4);
    float*  Gbuf  = (float*) alloc((size_t)2 * Bn * G4 * 4);        //  8.4 MB
    float*  Gc    = (float*) alloc((size_t)Bn * G4 * 4);            //  4.2 MB
    bf16_t* Henc  = (bf16_t*)alloc((size_t)2 * 2 * Bn * Hn * 2);    //  2.1 MB (pingpong x dir)
    float*  Cenc  = (float*) alloc((size_t)2 * Bn * Hn * 4);        //  2.1 MB
    bf16_t* Hd    = (bf16_t*)alloc((size_t)2 * Bn * Hn * 2);        //  1.0 MB (pingpong)
    float*  Cdec  = (float*) alloc((size_t)Bn * Hn * 4);            //  1.0 MB
    bf16_t* Hcat  = (bf16_t*)alloc((size_t)Bn * 2 * Hn * 2);        //  1.0 MB
    bf16_t* latb  = (bf16_t*)alloc((size_t)Bn * Ln * 2);
    float*  dhf   = (float*) alloc((size_t)Bn * Hn * 4);

    auto cdiv = [](size_t a, size_t b) { return (unsigned)((a + b - 1) / b); };

    // ---- pack weights to bf16 (L2-resident working set), fold biases ----
    pack_concat<<<cdiv((size_t)G4 * KE, 256), 256, 0, stream>>>(e_wih_f, e_whh_f, Wp, G4, En, Hn);
    pack_concat<<<cdiv((size_t)G4 * KE, 256), 256, 0, stream>>>(e_wih_b, e_whh_b, Wp + (size_t)G4 * KE, G4, En, Hn);
    pack_strided<<<cdiv((size_t)G4 * Hn, 256), 256, 0, stream>>>(d_whh, Wdhh, G4, Hn, Hn, 0, G4);
    pack_strided<<<cdiv((size_t)G4 * Hn, 256), 256, 0, stream>>>(d_wih, Wdih, G4, Hn, KE, En, G4);
    pack_strided<<<cdiv((size_t)Ln * 2 * Hn, 256), 256, 0, stream>>>(enc_w, Wenc, Ln, 2 * Hn, 2 * Hn, 0, Ln);
    pack_strided<<<cdiv((size_t)Hn * Ln, 256), 256, 0, stream>>>(dec_w, Wdec, Hn, Ln, Ln, 0, Hn);
    pack_strided<<<cdiv((size_t)128 * Hn, 256), 256, 0, stream>>>(out_w, Wout, 128, Hn, Hn, 0, Vn);
    add2<<<cdiv(G4, 256), 256, 0, stream>>>(e_bih_f, e_bhh_f, biasE, G4);
    add2<<<cdiv(G4, 256), 256, 0, stream>>>(e_bih_b, e_bhh_b, biasE + G4, G4);
    add2<<<cdiv(G4, 256), 256, 0, stream>>>(d_bih, d_bhh, biasD, G4);
    embed_gather<<<cdiv((size_t)Sn * Bn * En, 256), 256, 0, stream>>>(x, emb, Xb);
    zero_bf16<<<cdiv((size_t)2 * Bn * Hn, 256), 256, 0, stream>>>(Henc, (size_t)2 * Bn * Hn);
    zero_f32<<<cdiv((size_t)2 * Bn * Hn, 256), 256, 0, stream>>>(Cenc, (size_t)2 * Bn * Hn);
    zero_f32<<<cdiv((size_t)Bn * Hn, 256), 256, 0, stream>>>(Cdec, (size_t)Bn * Hn);

    // ---- bidirectional encoder: 128 sequential fused-K WMMA steps ----
    int p = 0;
    for (int t = 0; t < Sn; ++t) {
        enc_step_gemm<<<dim3(G4 / 128, Bn / 128, 2), 256, 0, stream>>>(
            Xb, Wp, Henc + (size_t)p * 2 * Bn * Hn, biasE, Gbuf, t);
        int q = p ^ 1;
        lstm_pointwise<<<dim3(Bn * Hn / 256, 2), 256, 0, stream>>>(
            Gbuf, (size_t)Bn * G4, Cenc, (size_t)Bn * Hn,
            Henc + (size_t)q * 2 * Bn * Hn, (size_t)Bn * Hn,
            (t == Sn - 1) ? Hcat : (bf16_t*)nullptr, (size_t)Hn, 2 * Hn);
        p = q;
    }

    // ---- latent = [hf,hb] @ enc_w^T + enc_b (into d_out tail, mirrored bf16) ----
    gemm_bf16_wmma<<<dim3(Ln / 128, Bn / 128), 256, 0, stream>>>(
        Hcat, 2 * Hn, Wenc, 2 * Hn,
        outF + (size_t)Bn * Sn * Vn, Ln, nullptr, enc_b, latb, Ln, 2 * Hn, Ln);
    // ---- dh = latent @ dec_w^T + dec_b ; bf16 mirror doubles as decoder h0 ----
    gemm_bf16_wmma<<<dim3(Hn / 128, Bn / 128), 256, 0, stream>>>(
        latb, Ln, Wdec, Ln, dhf, Hn, nullptr, dec_b, Hd, Hn, Ln, Hn);
    // ---- Gc = dh @ d_wih[:,E:]^T + (d_bih + d_bhh) (constant decoder input projection) ----
    gemm_bf16_wmma<<<dim3(G4 / 128, Bn / 128), 256, 0, stream>>>(
        Hd, Hn, Wdih, Hn, Gc, G4, nullptr, biasD, nullptr, 0, Hn, G4);

    // ---- decoder: 128 sequential steps, fused per-step vocab projection ----
    p = 0;
    for (int t = 0; t < Sn; ++t) {
        gemm_bf16_wmma<<<dim3(G4 / 128, Bn / 128), 256, 0, stream>>>(
            Hd + (size_t)p * Bn * Hn, Hn, Wdhh, Hn, Gbuf, G4, Gc, nullptr, nullptr, 0, Hn, G4);
        int q = p ^ 1;
        lstm_pointwise<<<dim3(Bn * Hn / 256, 1), 256, 0, stream>>>(
            Gbuf, 0, Cdec, 0, Hd + (size_t)q * Bn * Hn, 0, (bf16_t*)nullptr, 0, 0);
        // logits[:, t, :] = h_t @ out_w^T + out_b  (N padded to 128, stores guarded to 30)
        gemm_bf16_wmma<<<dim3(1, Bn / 128), 256, 0, stream>>>(
            Hd + (size_t)q * Bn * Hn, Hn, Wout, Hn,
            outF + (size_t)t * Vn, Sn * Vn, nullptr, out_b, nullptr, 0, Hn, Vn);
        p = q;
    }
}
